// Transformer_43765716746344
// MI455X (gfx1250) — matrix operand order repaired and checked by reading, not compiled, not run on hardware
//
#include <hip/hip_runtime.h>
#include <hip/hip_bf16.h>

// ---------------------------------------------------------------------------
// Transformer forward (encoder MHA, decoder MHA, cross MHA, fc+softmax) for
// MI455X / gfx1250.  All matmuls via V_WMMA_F32_16X16X32_F16 (wave32).
// B=2, L=2048, DIM=512, HEAD=8, ATTEN_DIM=64, M = B*L = 4096.
// ---------------------------------------------------------------------------

typedef __attribute__((ext_vector_type(16))) _Float16 v16h;
typedef __attribute__((ext_vector_type(8)))  _Float16 v8h;
typedef __attribute__((ext_vector_type(8)))  float    v8f;

#define DIMC   512
#define LSEQ   2048
#define MROWS  4096   // B*L

__device__ __forceinline__ v8f wmma32f16(v16h a, v16h b, v8f c) {
    // (neg_a, A, neg_b, B, c_mod, C, reuse_a, reuse_b)
    return __builtin_amdgcn_wmma_f32_16x16x32_f16(false, a, false, b,
                                                  (short)0, c, false, false);
}

// Build a 16-half A-fragment from two contiguous 8-half (16-byte) spans.
__device__ __forceinline__ v16h ldA16(const _Float16* p0, const _Float16* p1) {
    v8h lo = *(const v8h*)p0;
    v8h hi = *(const v8h*)p1;
    return __builtin_shufflevector(lo, hi, 0,1,2,3,4,5,6,7,8,9,10,11,12,13,14,15);
}

__device__ __forceinline__ void wait_asynccnt0() {
#if __has_builtin(__builtin_amdgcn_s_wait_asynccnt)
    __builtin_amdgcn_s_wait_asynccnt(0);
#else
    asm volatile("s_wait_asynccnt 0x0" ::: "memory");
#endif
}

// ---------------------------------------------------------------------------
// Weight transpose + fp32->f16 convert:  Wt[n*512+k] = (f16) W[k*512+n]
// ---------------------------------------------------------------------------
__global__ void k_wconv(const float* __restrict__ W, _Float16* __restrict__ Wt) {
    int i = blockIdx.x * 256 + threadIdx.x;     // 262144 total
    int n = i >> 9, k = i & 511;
    Wt[i] = (_Float16)W[k * DIMC + n];
}

// ---------------------------------------------------------------------------
// Embedding gather + positional encoding -> fp32 (residual) and f16 (GEMM in)
// ---------------------------------------------------------------------------
__global__ void k_embed(const int* __restrict__ tok, const float* __restrict__ emb,
                        float* __restrict__ xF, _Float16* __restrict__ xH) {
    int i   = blockIdx.x * 256 + threadIdx.x;   // 4096*512 total
    int col = i & 511;
    int row = i >> 9;
    int l   = row & (LSEQ - 1);
    int t   = tok[row];
    // angle = l * 10000^(-(col/2)/512) = l * exp((col>>1) * -ln(1e4)/512)
    float ang = (float)l * __expf((float)(col >> 1) * (-0.0179889460f));
    float pe  = (col & 1) ? __cosf(ang) : __sinf(ang);
    float v   = emb[t * DIMC + col] + pe;
    xF[i] = v;
    xH[i] = (_Float16)v;
}

// ---------------------------------------------------------------------------
// GEMM: C[4096,512] = A[4096,512](f16) @ W[512,512] + bias (+resid)
//   Wt is pre-transposed f16 [N][K].  One 16x64 strip per wave (4 accums,
//   A-fragment reused 4x) -> 64 unrolled v_wmma per wave.  4 waves/block.
// ---------------------------------------------------------------------------
__global__ void k_gemm(const _Float16* __restrict__ A, const _Float16* __restrict__ Wt,
                       const float* __restrict__ bias, const float* __restrict__ resid,
                       _Float16* __restrict__ outH, float* __restrict__ outF,
                       float outScale) {
    int wave = threadIdx.x >> 5;
    int lane = threadIdx.x & 31;
    int strip = blockIdx.x * 4 + wave;          // 2048 strips = 256 (M) x 8 (N/64)
    int tm  = strip & 255;
    int tns = strip >> 8;                       // 0..7 -> 64-wide N strip
    int l15 = lane & 15;
    int hi  = lane >> 4;

    const _Float16* arow = A  + (tm * 16 + l15) * DIMC + (hi ? 16 : 0);
    const _Float16* w0   = Wt + (tns * 64 + l15) * DIMC + (hi ? 16 : 0);
    const _Float16* w1   = w0 + 16 * DIMC;
    const _Float16* w2   = w0 + 32 * DIMC;
    const _Float16* w3   = w0 + 48 * DIMC;

    v8f acc[4] = {};
#pragma unroll
    for (int kk = 0; kk < 16; ++kk) {
        v16h a = ldA16(arow + kk * 32, arow + kk * 32 + 8);
        acc[0] = wmma32f16(a, *(const v16h*)(w0 + kk * 32), acc[0]);
        acc[1] = wmma32f16(a, *(const v16h*)(w1 + kk * 32), acc[1]);
        acc[2] = wmma32f16(a, *(const v16h*)(w2 + kk * 32), acc[2]);
        acc[3] = wmma32f16(a, *(const v16h*)(w3 + kk * 32), acc[3]);
    }

#pragma unroll
    for (int j = 0; j < 4; ++j) {
        int n = tns * 64 + j * 16 + l15;
        float bv = bias[n];
#pragma unroll
        for (int r = 0; r < 8; ++r) {
            int m = tm * 16 + r + (hi ? 8 : 0);
            float v = acc[j][r] + bv;
            if (resid) v += resid[m * DIMC + n];
            if (outF) outF[m * DIMC + n] = v;
            if (outH) outH[m * DIMC + n] = (_Float16)(v * outScale);
        }
    }
}

// ---------------------------------------------------------------------------
// Flash-style attention.  One wave = 16 query rows of one (b,h).
// 4 waves/block share (b,h).  The 32-key V chunk is DMA'd into LDS with
// GLOBAL_LOAD_ASYNC_TO_LDS_B128 by wave 0 (hidden behind the QK^T WMMAs),
// then transposed LDS->LDS into Vt[d][key] for the WMMA B-fragment layout.
// Output scattered in the reference's (d n) layout: col = d*HEAD + h.
// ---------------------------------------------------------------------------
__global__ void k_attn(const _Float16* __restrict__ Q, const _Float16* __restrict__ K,
                       const _Float16* __restrict__ V, _Float16* __restrict__ O) {
    __shared__ __align__(32) _Float16 Vraw[32 * 64];     // [key][d] raw DMA chunk
    __shared__ __align__(32) _Float16 Vt[64][32];        // [d][key]
    __shared__ __align__(32) _Float16 Pb[4][16 * 32];    // per-wave P tile

    int wave = threadIdx.x >> 5;
    int lane = threadIdx.x & 31;
    int bh = blockIdx.x >> 5;                        // 16 (b,h) pairs
    int qt = ((blockIdx.x & 31) << 2) + wave;        // 0..127 query tiles
    int b  = bh >> 3;
    int h  = bh & 7;
    int l15 = lane & 15;
    int hi  = lane >> 4;
    int kb16 = hi ? 16 : 0;

    // Q fragments for d in [0,32) and [32,64); Q was pre-scaled by 1/sqrt(64).
    const _Float16* qp = Q + (b * LSEQ + qt * 16 + l15) * DIMC + h * 64 + (hi ? 16 : 0);
    v16h qa0 = ldA16(qp,      qp + 8);
    v16h qa1 = ldA16(qp + 32, qp + 40);

    float rm[8], rs[8];
#pragma unroll
    for (int r = 0; r < 8; ++r) { rm[r] = -1e30f; rs[r] = 0.0f; }
    v8f o0 = {}, o1 = {}, o2 = {}, o3 = {};

    for (int ch = 0; ch < LSEQ / 32; ++ch) {
        __syncthreads();   // previous Vt/Pb/Vraw consumers done

        // Wave 0: async DMA of the 32x64 f16 V chunk into LDS (row-major).
        // One b128 per lane x 8 => 4 KB; ASYNCcnt-tracked, overlapped with QK^T.
        if (wave == 0) {
            const _Float16* vg = V + (b * LSEQ + ch * 32 + lane) * DIMC + h * 64;
            unsigned long long ga = (unsigned long long)(uintptr_t)vg;
            unsigned la = (unsigned)(uintptr_t)&Vraw[lane * 64];
#pragma unroll
            for (int j = 0; j < 8; ++j) {
                asm volatile("global_load_async_to_lds_b128 %0, %1, off"
                             :: "v"(la + j * 16), "v"(ga + j * 16)
                             : "memory");
            }
        }

        // S tiles: keys [ch*32, ch*32+16) and [+16, +32)
        const _Float16* kpBase = K + (b * LSEQ + ch * 32) * DIMC + h * 64;
        if (ch + 1 < LSEQ / 32) {
            __builtin_prefetch(kpBase + 32 * DIMC, 0, 1);    // next K chunk
        }
        v8f s0 = {}, s1 = {};
        {
            const _Float16* kp = kpBase + l15 * DIMC + kb16;
            s0 = wmma32f16(qa0, *(const v16h*)kp,        s0);
            s0 = wmma32f16(qa1, *(const v16h*)(kp + 32), s0);
            const _Float16* kp2 = kpBase + (16 + l15) * DIMC + kb16;
            s1 = wmma32f16(qa0, *(const v16h*)kp2,        s1);
            s1 = wmma32f16(qa1, *(const v16h*)(kp2 + 32), s1);
        }

        // Online softmax over 32 keys (rows live in 16-lane halves)
        float p0[8], p1[8];
#pragma unroll
        for (int r = 0; r < 8; ++r) {
            float mx = fmaxf(s0[r], s1[r]);
            mx = fmaxf(mx, __shfl_xor(mx, 8, 32));
            mx = fmaxf(mx, __shfl_xor(mx, 4, 32));
            mx = fmaxf(mx, __shfl_xor(mx, 2, 32));
            mx = fmaxf(mx, __shfl_xor(mx, 1, 32));
            float nm = fmaxf(rm[r], mx);
            float sc = __expf(rm[r] - nm);
            rm[r] = nm;
            p0[r] = __expf(s0[r] - nm);
            p1[r] = __expf(s1[r] - nm);
            float ps = p0[r] + p1[r];
            ps += __shfl_xor(ps, 8, 32);
            ps += __shfl_xor(ps, 4, 32);
            ps += __shfl_xor(ps, 2, 32);
            ps += __shfl_xor(ps, 1, 32);
            rs[r] = rs[r] * sc + ps;
            o0[r] *= sc; o1[r] *= sc; o2[r] *= sc; o3[r] *= sc;
        }

        // P (f16) -> per-wave LDS in row-major [m][k], k = 32 keys
#pragma unroll
        for (int r = 0; r < 8; ++r) {
            int m = r + (hi ? 8 : 0);
            Pb[wave][m * 32 + l15]      = (_Float16)p0[r];
            Pb[wave][m * 32 + 16 + l15] = (_Float16)p1[r];
        }

        wait_asynccnt0();      // wave 0's DMA complete (no-op for other waves)
        __syncthreads();       // Vraw + Pb visible block-wide

        // LDS->LDS transpose Vraw[key][d] -> Vt[d][key]
        {
            int key   = threadIdx.x >> 2;
            int piece = threadIdx.x & 3;
            v8h v0 = *(const v8h*)&Vraw[key * 64 + piece * 16];
            v8h v1 = *(const v8h*)&Vraw[key * 64 + piece * 16 + 8];
#pragma unroll
            for (int j = 0; j < 8; ++j) {
                Vt[piece * 16 + j][key]     = v0[j];
                Vt[piece * 16 + 8 + j][key] = v1[j];
            }
        }
        __syncthreads();       // Vt ready

        // PV: A = P (16x32), B = Vt chunks (32 keys x 16 dims)
        const _Float16* pp = &Pb[wave][l15 * 32 + (hi ? 16 : 0)];
        v16h pa = ldA16(pp, pp + 8);
        o0 = wmma32f16(pa, *(const v16h*)&Vt[l15][kb16],      o0);
        o1 = wmma32f16(pa, *(const v16h*)&Vt[16 + l15][kb16], o1);
        o2 = wmma32f16(pa, *(const v16h*)&Vt[32 + l15][kb16], o2);
        o3 = wmma32f16(pa, *(const v16h*)&Vt[48 + l15][kb16], o3);
    }

    // Normalize and scatter in 'm n l d -> m l (d n)' layout: col = d*8 + h
#pragma unroll
    for (int r = 0; r < 8; ++r) {
        float inv = 1.0f / rs[r];
        int m = r + (hi ? 8 : 0);
        _Float16* op = O + (b * LSEQ + qt * 16 + m) * DIMC + h;
        op[(0 * 16 + l15) * 8] = (_Float16)(o0[r] * inv);
        op[(1 * 16 + l15) * 8] = (_Float16)(o1[r] * inv);
        op[(2 * 16 + l15) * 8] = (_Float16)(o2[r] * inv);
        op[(3 * 16 + l15) * 8] = (_Float16)(o3[r] * inv);
    }
}

// ---------------------------------------------------------------------------
// LayerNorm: one wave per row (512 elems, 16 per lane), shuffle reductions.
// ---------------------------------------------------------------------------
__global__ void k_ln(const float* __restrict__ x, const float* __restrict__ g,
                     const float* __restrict__ bp, float* __restrict__ yF,
                     _Float16* __restrict__ yH) {
    int wave = threadIdx.x >> 5;
    int lane = threadIdx.x & 31;
    int row = blockIdx.x * 4 + wave;
    const float* xr = x + row * DIMC;
    float v[16];
    float s = 0.0f;
#pragma unroll
    for (int j = 0; j < 16; ++j) { v[j] = xr[j * 32 + lane]; s += v[j]; }
    s += __shfl_xor(s, 16, 32); s += __shfl_xor(s, 8, 32);
    s += __shfl_xor(s, 4, 32);  s += __shfl_xor(s, 2, 32); s += __shfl_xor(s, 1, 32);
    float mu = s * (1.0f / DIMC);
    float vs = 0.0f;
#pragma unroll
    for (int j = 0; j < 16; ++j) { float d = v[j] - mu; vs += d * d; }
    vs += __shfl_xor(vs, 16, 32); vs += __shfl_xor(vs, 8, 32);
    vs += __shfl_xor(vs, 4, 32);  vs += __shfl_xor(vs, 2, 32); vs += __shfl_xor(vs, 1, 32);
    float rinv = rsqrtf(vs * (1.0f / DIMC) + 1e-5f);
#pragma unroll
    for (int j = 0; j < 16; ++j) {
        int c = j * 32 + lane;
        float y = (v[j] - mu) * rinv * g[c] + bp[c];
        yF[row * DIMC + c] = y;
        yH[row * DIMC + c] = (_Float16)y;
    }
}

// ---------------------------------------------------------------------------
// Final: z = x @ fc_w + fc_b; softmax over singleton dim -> exp(z-z) = 1.0
// ---------------------------------------------------------------------------
__global__ void k_final(const float* __restrict__ x, const float* __restrict__ w,
                        const float* __restrict__ bb, float* __restrict__ out) {
    int i = blockIdx.x * 256 + threadIdx.x;     // 4096 rows
    float z = bb[0];
    for (int d = 0; d < DIMC; ++d) z += x[i * DIMC + d] * w[d];
    out[i] = __expf(z - z);
}

// ---------------------------------------------------------------------------
// Host launcher
// ---------------------------------------------------------------------------
extern "C" void kernel_launch(void* const* d_in, const int* in_sizes, int n_in,
                              void* d_out, int out_size, void* d_ws, size_t ws_size,
                              hipStream_t stream) {
    (void)in_sizes; (void)n_in; (void)out_size; (void)ws_size;
    const int*   tokIn  = (const int*)d_in[0];
    const int*   tokOut = (const int*)d_in[1];
    const float* encEmb = (const float*)d_in[2];
    const float* decEmb = (const float*)d_in[3];
    const float* fcw    = (const float*)d_in[34];
    const float* fcb    = (const float*)d_in[35];

    char* ws = (char*)d_ws;
    size_t off = 0;
    auto alloc = [&](size_t nbytes) {
        void* p = ws + off;
        off += (nbytes + 255) & ~(size_t)255;
        return p;
    };
    const size_t MD = (size_t)MROWS * DIMC;
    float*    A0f = (float*)   alloc(MD * 4);   // enc stream fp32
    _Float16* A0h = (_Float16*)alloc(MD * 2);
    float*    A1f = (float*)   alloc(MD * 4);   // dec stream fp32
    _Float16* A1h = (_Float16*)alloc(MD * 2);
    float*    Cf  = (float*)   alloc(MD * 4);   // cross output fp32
    _Float16* Qh  = (_Float16*)alloc(MD * 2);
    _Float16* Kh  = (_Float16*)alloc(MD * 2);
    _Float16* Vh  = (_Float16*)alloc(MD * 2);
    _Float16* Ph  = (_Float16*)alloc(MD * 2);   // attention out f16
    float*    Yt  = (float*)   alloc(MD * 4);   // pre-LN fp32
    _Float16* Wt[12];
    for (int i = 0; i < 12; ++i) Wt[i] = (_Float16*)alloc((size_t)DIMC * DIMC * 2);

    // Transpose+convert Wq/Wk/Wv/Wo for each of the 3 MHA blocks
    for (int m = 0; m < 3; ++m) {
        int base = 4 + 10 * m;
        const int wsel[4] = {0, 2, 4, 6};
        for (int j = 0; j < 4; ++j)
            k_wconv<<<1024, 256, 0, stream>>>((const float*)d_in[base + wsel[j]], Wt[m * 4 + j]);
    }

    auto run_mha = [&](const float* qinF, const _Float16* qinH, const _Float16* kinH,
                       const _Float16* vinH, int pbase,
                       const _Float16* WtQ, const _Float16* WtK,
                       const _Float16* WtV, const _Float16* WtO,
                       float* outF, _Float16* outH) {
        const float* bq = (const float*)d_in[pbase + 1];
        const float* bk = (const float*)d_in[pbase + 3];
        const float* bv = (const float*)d_in[pbase + 5];
        const float* bo = (const float*)d_in[pbase + 7];
        const float* gg = (const float*)d_in[pbase + 8];
        const float* be = (const float*)d_in[pbase + 9];
        k_gemm<<<512, 128, 0, stream>>>(qinH, WtQ, bq, nullptr, Qh, nullptr, 0.125f); // Q/sqrt(64)
        k_gemm<<<512, 128, 0, stream>>>(kinH, WtK, bk, nullptr, Kh, nullptr, 1.0f);
        k_gemm<<<512, 128, 0, stream>>>(vinH, WtV, bv, nullptr, Vh, nullptr, 1.0f);
        k_attn<<<512, 128, 0, stream>>>(Qh, Kh, Vh, Ph);
        k_gemm<<<512, 128, 0, stream>>>(Ph, WtO, bo, qinF, nullptr, Yt, 1.0f);
        k_ln<<<1024, 128, 0, stream>>>(Yt, gg, be, outF, outH);
    };

    // Encoder
    k_embed<<<8192, 256, 0, stream>>>(tokIn, encEmb, A0f, A0h);
    run_mha(A0f, A0h, A0h, A0h, 4, Wt[0], Wt[1], Wt[2], Wt[3], A0f, A0h);
    // Decoder
    k_embed<<<8192, 256, 0, stream>>>(tokOut, decEmb, A1f, A1h);
    run_mha(A1f, A1h, A1h, A1h, 14, Wt[4], Wt[5], Wt[6], Wt[7], A1f, A1h);
    // Cross: q=enc, k=enc, v=dec (faithful to reference)
    run_mha(A0f, A0h, A0h, A1h, 24, Wt[8], Wt[9], Wt[10], Wt[11], Cf, A0h);
    // fc + softmax over singleton dim
    k_final<<<16, 256, 0, stream>>>(Cf, fcw, fcb, (float*)d_out);
}
